// GNNEmbedder_25417616458217
// MI455X (gfx1250) — matrix operand order repaired and checked by reading, not compiled, not run on hardware
//
#include <hip/hip_runtime.h>
#include <hip/hip_bf16.h>

// ---------------------------------------------------------------------------
// GIN embedder for MI455X (gfx1250, wave32).
//   - edge scatter-add: float4 atomics, 16 lanes/edge (memory-bound phase)
//   - per-layer fused: (h+agg) @ W1 + b1 -> BN(eval) -> ReLU -> @ W2 + b2 -> ReLU
//     using v_wmma_f32_16x16x32_f16 (f16 operands, f32 accumulate)
//   - MLP reuses the same fused kernel (no BN, no final ReLU)
//   - graph pooling: float4 atomics into 128x64 output
// ---------------------------------------------------------------------------

typedef __attribute__((ext_vector_type(16))) _Float16 v16h;
typedef __attribute__((ext_vector_type(8)))  float    v8f;

union V16H { v16h v; unsigned u[8]; };

#define D 64
#define ROWS_PER_WAVE 16
#define WAVES_PER_BLOCK 4
#define ROWS_PER_BLOCK (ROWS_PER_WAVE * WAVES_PER_BLOCK)

// ---- A operand gather: 16x64 row-major f16 tile in LDS -> v16h for one wave.
// ISA 7.12.2 (16-bit A 16x32): lane<16 row=lane holds K = kbase+{0..7, 16..23},
// lane>=16 row=lane-16 holds K = kbase+{8..15, 24..31}. Pairs of K are packed
// per VGPR -> contiguous uint loads from the row-major tile.
__device__ inline v16h gather_a(const _Float16* tile, int lane, int kbase) {
    int m  = lane & 15;
    int hi = (lane >> 4) & 1;
    const unsigned* t32 = reinterpret_cast<const unsigned*>(tile);
    int b0 = (m * D + kbase + 8 * hi) >> 1;   // uint index; 8 contiguous halves
    V16H r;
#pragma unroll
    for (int i = 0; i < 4; ++i) r.u[i]     = t32[b0 + i];        // K = kbase+8hi+0..7
#pragma unroll
    for (int i = 0; i < 4; ++i) r.u[4 + i] = t32[b0 + 8 + i];    // K = +16
    return r.v;
}

// ---- B operand gather from transposed weight tile Wt[n][k] (64x64 f16 in LDS).
// ISA 7.12.2 (B 32x16): lane<16 col=lane holds K = kbase+{0..7, 16..23} (pairs per VGPR),
// lane>=16 col=lane-16 holds K = kbase+{8..15, 24..31} (same order as gather_a).
__device__ inline v16h gather_b(const _Float16* wt, int lane, int ntile, int kbase) {
    int n  = ntile * 16 + (lane & 15);
    int kb = kbase + 8 * ((lane >> 4) & 1);
    const unsigned* t32 = reinterpret_cast<const unsigned*>(wt + n * D + kb);
    V16H r;
#pragma unroll
    for (int i = 0; i < 8; ++i) r.u[i] = t32[(i & 3) + 8 * (i >> 2)];
    return r.v;
}

__device__ inline v8f wmma_f16(v16h a, v16h b, v8f c) {
    return __builtin_amdgcn_wmma_f32_16x16x32_f16(
        /*neg_a=*/false, a, /*neg_b=*/false, b,
        /*c_mod=*/(short)0, c, /*reuse_a=*/false, /*reuse_b=*/false);
}

// ---------------------------------------------------------------------------
__global__ void fill_zero_kernel(float* __restrict__ p, int n) {
    int i = blockIdx.x * blockDim.x + threadIdx.x;
    if (i < n) p[i] = 0.0f;
}

// ---------------------------------------------------------------------------
// agg[dst] += h[src], 16 lanes per edge, float4 per lane.
__global__ void scatter_agg_kernel(const float* __restrict__ h,
                                   const int* __restrict__ src,
                                   const int* __restrict__ dst,
                                   float* __restrict__ agg, int E) {
    int t = blockIdx.x * blockDim.x + threadIdx.x;
    int e = t >> 4;
    if (e >= E) return;
    int c = (t & 15) * 4;
    int s = src[e];
    int d = dst[e];
    const float4 v = *reinterpret_cast<const float4*>(h + (size_t)s * D + c);
    float* o = agg + (size_t)d * D + c;
    atomicAdd(o + 0, v.x);
    atomicAdd(o + 1, v.y);
    atomicAdd(o + 2, v.z);
    atomicAdd(o + 3, v.w);
}

// ---------------------------------------------------------------------------
// Fused: z = act1((hin[+agg]) @ W1 + b1 [BN]) ; hout = act2(z @ W2 + b2)
// W1/W2 are [k][n] row-major in global; transposed to [n][k] f16 in LDS.
__global__ __launch_bounds__(128) void gin_fused_kernel(
    const float* __restrict__ hin, const float* __restrict__ agg,
    const float* __restrict__ W1,  const float* __restrict__ b1,
    const float* __restrict__ gamma, const float* __restrict__ beta,
    const float* __restrict__ rmean, const float* __restrict__ rvar,
    const float* __restrict__ W2,  const float* __restrict__ b2,
    float* __restrict__ hout, int N, int relu2) {

    __shared__ _Float16 sW1t[D * D];                       // [n][k]  8 KB
    __shared__ _Float16 sW2t[D * D];                       // [n][k]  8 KB
    __shared__ _Float16 sX[WAVES_PER_BLOCK][ROWS_PER_WAVE * D];  // 8 KB
    __shared__ _Float16 sZ[WAVES_PER_BLOCK][ROWS_PER_WAVE * D];  // 8 KB

    const int tid  = threadIdx.x;
    const int w    = tid >> 5;
    const int lane = tid & 31;
    const int r0   = blockIdx.x * ROWS_PER_BLOCK;
    const int r0w  = r0 + w * ROWS_PER_WAVE;
    if (r0 >= N) return;

    // Stage weights (transposed) into LDS, f32 -> f16.
#pragma unroll
    for (int idx = tid; idx < D * D; idx += 128) {
        int k = idx >> 6, n = idx & 63;
        sW1t[n * D + k] = (_Float16)W1[idx];
        sW2t[n * D + k] = (_Float16)W2[idx];
    }

    // Stage this wave's 16x64 input tile (h + agg) as f16.
#pragma unroll
    for (int j = 0; j < 8; ++j) {
        int off = j * 128 + lane * 4;              // element offset in tile
        int m = off >> 6, k = off & 63;
        int row = r0w + m; if (row >= N) row = N - 1;
        float4 v = *reinterpret_cast<const float4*>(hin + (size_t)row * D + k);
        if (agg) {
            float4 a = *reinterpret_cast<const float4*>(agg + (size_t)row * D + k);
            v.x += a.x; v.y += a.y; v.z += a.z; v.w += a.w;
        }
        sX[w][off + 0] = (_Float16)v.x;
        sX[w][off + 1] = (_Float16)v.y;
        sX[w][off + 2] = (_Float16)v.z;
        sX[w][off + 3] = (_Float16)v.w;
    }
    __syncthreads();

    const int col_lo = lane & 15;
    const int mbase  = 8 * (lane >> 4);

    // ---- GEMM1 + bias + BN(eval) + ReLU -> sZ (f16)
    {
        v16h a0 = gather_a(sX[w], lane, 0);
        v16h a1 = gather_a(sX[w], lane, 32);
#pragma unroll
        for (int nt = 0; nt < 4; ++nt) {
            int col = nt * 16 + col_lo;
            v8f c = {};
            c = wmma_f16(a0, gather_b(sW1t, lane, nt, 0),  c);
            c = wmma_f16(a1, gather_b(sW1t, lane, nt, 32), c);
            float bb = b1[col];
            float sc = 1.0f, sh = 0.0f;
            if (gamma) {
                float inv = rsqrtf(rvar[col] + 1e-5f);
                sc = gamma[col] * inv;
                sh = beta[col] - rmean[col] * sc;
            }
#pragma unroll
            for (int j = 0; j < 8; ++j) {
                float z = c[j] + bb;
                z = z * sc + sh;
                z = fmaxf(z, 0.0f);
                sZ[w][(j + mbase) * D + col] = (_Float16)z;
            }
        }
    }
    __syncthreads();

    // ---- GEMM2 + bias (+ReLU) -> global
    {
        v16h a0 = gather_a(sZ[w], lane, 0);
        v16h a1 = gather_a(sZ[w], lane, 32);
#pragma unroll
        for (int nt = 0; nt < 4; ++nt) {
            int col = nt * 16 + col_lo;
            v8f c = {};
            c = wmma_f16(a0, gather_b(sW2t, lane, nt, 0),  c);
            c = wmma_f16(a1, gather_b(sW2t, lane, nt, 32), c);
            float bb = b2[col];
#pragma unroll
            for (int j = 0; j < 8; ++j) {
                float z = c[j] + bb;
                if (relu2) z = fmaxf(z, 0.0f);
                int row = r0w + j + mbase;
                if (row < N) hout[(size_t)row * D + col] = z;
            }
        }
    }
}

// ---------------------------------------------------------------------------
// out[batch[n]] += h[n], 16 lanes per node, float4 per lane.
__global__ void pool_kernel(const float* __restrict__ h,
                            const int* __restrict__ batch,
                            float* __restrict__ out, int N) {
    int t = blockIdx.x * blockDim.x + threadIdx.x;
    int n = t >> 4;
    if (n >= N) return;
    int c = (t & 15) * 4;
    int g = batch[n];
    const float4 v = *reinterpret_cast<const float4*>(h + (size_t)n * D + c);
    float* o = out + (size_t)g * D + c;
    atomicAdd(o + 0, v.x);
    atomicAdd(o + 1, v.y);
    atomicAdd(o + 2, v.z);
    atomicAdd(o + 3, v.w);
}

// ---------------------------------------------------------------------------
extern "C" void kernel_launch(void* const* d_in, const int* in_sizes, int n_in,
                              void* d_out, int out_size, void* d_ws, size_t ws_size,
                              hipStream_t stream) {
    const float* x     = (const float*)d_in[0];
    const int*   ei    = (const int*)d_in[1];    // [2, E]: src then dst
    const int*   bid   = (const int*)d_in[2];
    const float* W1    = (const float*)d_in[3];  // [L, 64, 64]
    const float* b1    = (const float*)d_in[4];
    const float* gamma = (const float*)d_in[5];
    const float* beta  = (const float*)d_in[6];
    const float* rmean = (const float*)d_in[7];
    const float* rvar  = (const float*)d_in[8];
    const float* W2    = (const float*)d_in[9];
    const float* b2    = (const float*)d_in[10];
    const float* Wm1   = (const float*)d_in[11];
    const float* bm1   = (const float*)d_in[12];
    const float* Wm2   = (const float*)d_in[13];
    const float* bm2   = (const float*)d_in[14];

    const int N = in_sizes[0] / D;
    const int E = in_sizes[1] / 2;
    const int L = in_sizes[3] / (D * D);

    float* agg = (float*)d_ws;
    float* hA  = agg + (size_t)N * D;
    float* hB  = hA  + (size_t)N * D;

    const int nd = N * D;
    const int fz_blocks = (nd + 255) / 256;
    const int sc_blocks = (int)(((long long)E * 16 + 255) / 256);
    const int gemm_blocks = (N + ROWS_PER_BLOCK - 1) / ROWS_PER_BLOCK;

    const float* cur = x;
    float* nxt = hA;
    for (int l = 0; l < L; ++l) {
        fill_zero_kernel<<<fz_blocks, 256, 0, stream>>>(agg, nd);
        scatter_agg_kernel<<<sc_blocks, 256, 0, stream>>>(cur, ei, ei + E, agg, E);
        gin_fused_kernel<<<gemm_blocks, 128, 0, stream>>>(
            cur, agg,
            W1 + (size_t)l * D * D, b1 + (size_t)l * D,
            gamma + (size_t)l * D, beta + (size_t)l * D,
            rmean + (size_t)l * D, rvar + (size_t)l * D,
            W2 + (size_t)l * D * D, b2 + (size_t)l * D,
            nxt, N, /*relu2=*/1);
        cur = nxt;
        nxt = (nxt == hA) ? hB : hA;
    }

    // Downstream MLP: relu(h @ Wm1 + bm1) @ Wm2 + bm2 (no BN, no final ReLU)
    gin_fused_kernel<<<gemm_blocks, 128, 0, stream>>>(
        cur, nullptr, Wm1, bm1, nullptr, nullptr, nullptr, nullptr,
        Wm2, bm2, nxt, N, /*relu2=*/0);
    cur = nxt;

    fill_zero_kernel<<<(out_size + 255) / 256, 256, 0, stream>>>((float*)d_out, out_size);
    pool_kernel<<<(int)(((long long)N * 16 + 255) / 256), 256, 0, stream>>>(
        cur, bid, (float*)d_out, N);
}
